// HANLayer_1906965479534
// MI455X (gfx1250) — matrix operand order repaired and checked by reading, not compiled, not run on hardware
//
#include <hip/hip_runtime.h>
#include <hip/hip_bf16.h>
#include <math.h>

#define MPATH 3
#define DIN   256
#define KH    8
#define FD    32
#define CF    256   // KH*FD
#define HID   128
#define SLOPE 0.2f
#define SAPAD (DIN + 8)   // padded LDS row stride: 528B == 4 banks mod 64 -> conflict-free

typedef __attribute__((ext_vector_type(16))) _Float16 v16h;
typedef __attribute__((ext_vector_type(8)))  _Float16 v8h;
typedef __attribute__((ext_vector_type(8)))  float    v8f;

// ---------------- WMMA helpers ----------------

__device__ __forceinline__ v16h ld_frag(const _Float16* p0, const _Float16* p1) {
  // two 16-byte contiguous chunks -> one 16-half fragment register block
  v8h a = *(const v8h*)p0;
  v8h b = *(const v8h*)p1;
  v16h r;
#pragma unroll
  for (int i = 0; i < 8; ++i) { r[i] = a[i]; r[i + 8] = b[i]; }
  return r;
}

__device__ __forceinline__ v8f wmma32f16(v16h a, v16h b, v8f c) {
  // (neg_a, A, neg_b, B, c_mod, C, reuse_a, reuse_b)
  return __builtin_amdgcn_wmma_f32_16x16x32_f16(false, a, false, b, (short)0, c,
                                                false, false);
}

__device__ __forceinline__ void atomicMaxF(float* addr, float v) {
  // correct with *addr initialized to -inf; works across sign mixes
  if (!(v < 0.0f)) atomicMax((int*)addr, __float_as_int(v));
  else             atomicMin((unsigned int*)addr, __float_as_uint(v));
}

// cooperative 16xDIN f16 tile copy: global (stride DIN) -> LDS (stride SAPAD)
__device__ __forceinline__ void stage_a_tile(const _Float16* __restrict__ gA,
                                             _Float16* __restrict__ sA, int tid) {
  const int row = tid >> 4;          // 0..15
  const int seg = tid & 15;          // 0..15, 16 halves each
  const _Float16* src = gA + (size_t)row * DIN + seg * 16;
  _Float16*       dstp = sA + row * SAPAD + seg * 16;
  *(v8h*)dstp       = *(const v8h*)src;
  *(v8h*)(dstp + 8) = *(const v8h*)(src + 8);
}

// ---------------- prep / init ----------------

__global__ void k_cvt_h(const float* __restrict__ h, _Float16* __restrict__ h16,
                        int total) {
  int i = blockIdx.x * blockDim.x + threadIdx.x;
  if (i < total) h16[i] = (_Float16)h[i];
}

// W (M, DIN, CF) -> wT16 (M, CF, DIN)  (B columns become contiguous rows)
__global__ void k_cvt_w(const float* __restrict__ W, _Float16* __restrict__ wT16,
                        int total) {
  int i = blockIdx.x * blockDim.x + threadIdx.x;
  if (i >= total) return;
  int m   = i / (CF * DIN);
  int rem = i % (CF * DIN);
  int kf  = rem / DIN;
  int d   = rem % DIN;
  wT16[i] = (_Float16)W[((size_t)m * DIN + d) * CF + kf];
}

// W1 (DIN, HID) -> w1T16 (HID, DIN)
__global__ void k_cvt_w1(const float* __restrict__ W1, _Float16* __restrict__ w1T16,
                         int total) {
  int i = blockIdx.x * blockDim.x + threadIdx.x;
  if (i >= total) return;
  int c = i / DIN;
  int d = i % DIN;
  w1T16[i] = (_Float16)W1[(size_t)d * HID + c];
}

__global__ void k_init(float* __restrict__ emax, float* __restrict__ denom,
                       float* __restrict__ outAgg, float* __restrict__ wsum,
                       int mnk, int mnc) {
  int i = blockIdx.x * blockDim.x + threadIdx.x;
  if (i < mnk) { emax[i] = -__builtin_huge_valf(); denom[i] = 0.0f; }
  if (i < mnc) outAgg[i] = 0.0f;
  if (i < MPATH) wsum[i] = 0.0f;
}

// ---------------- projection GEMM: z[m] = h @ W_m (WMMA f16->f32) ----------------
// block = 256 threads = 8 waves; each block: 16 rows x 256 cols.
// A tile (16x256, 8KB) staged once in LDS (shared by all 8 waves);
// each wave: two 16x16 column tiles, K-loop 8 x (K=32), B streamed from global.

__global__ __launch_bounds__(256) void k_proj_gemm(
    const _Float16* __restrict__ h16, const _Float16* __restrict__ wT16,
    float* __restrict__ z, int N) {
  __shared__ _Float16 sA[16 * SAPAD];

  const int m       = blockIdx.y;
  const int rowBase = blockIdx.x * 16;          // N % 16 == 0 (10000/16 = 625)
  const int wave    = threadIdx.x >> 5;
  const int lane    = threadIdx.x & 31;
  const int r       = lane & 15;
  const int hs      = lane >> 4;

  stage_a_tile(h16 + (size_t)rowBase * DIN, sA, threadIdx.x);
  __syncthreads();

  const _Float16* A  = sA + r * SAPAD;          // LDS, conflict-free via SAPAD
  const _Float16* B  = wT16 + (size_t)m * CF * DIN;
  const int c0 = wave * 32, c1 = c0 + 16;
  const _Float16* B0 = B + (size_t)(c0 + r) * DIN + hs * 16;
  const _Float16* B1 = B + (size_t)(c1 + r) * DIN + hs * 16;

  v8f acc0 = {}, acc1 = {};
#pragma unroll
  for (int kk = 0; kk < DIN / 32; ++kk) {
    const int kb = kk * 32;
    if (kk + 2 < DIN / 32) {
      __builtin_prefetch(B0 + kb + 64, 0, 1);   // global_prefetch_b8
      __builtin_prefetch(B1 + kb + 64, 0, 1);
    }
    v16h a  = ld_frag(A + kb + hs * 16, A + kb + hs * 16 + 8);   // ds_load
    v16h b0 = ld_frag(B0 + kb, B0 + kb + 8);
    v16h b1 = ld_frag(B1 + kb, B1 + kb + 8);
    acc0 = wmma32f16(a, b0, acc0);
    acc1 = wmma32f16(a, b1, acc1);
  }
  float* zm = z + (size_t)m * N * CF;
#pragma unroll
  for (int j = 0; j < 8; ++j) {
    int row = rowBase + j + 8 * hs;             // C layout: VGPR j -> M=j / M=8+j
    zm[(size_t)row * CF + c0 + r] = acc0[j];
    zm[(size_t)row * CF + c1 + r] = acc1[j];
  }
}

// ---------------- per-node attention logits ----------------

__global__ void k_eler(const float* __restrict__ z, const float* __restrict__ al,
                       const float* __restrict__ ar, float* __restrict__ el,
                       float* __restrict__ er, int N, int total) {
  int i = blockIdx.x * blockDim.x + threadIdx.x;
  if (i >= total) return;
  int k  = i & (KH - 1);
  int mn = i >> 3;
  int m  = mn / N;
  const float* zr  = z + (size_t)mn * CF + k * FD;
  const float* alr = al + (size_t)m * CF + k * FD;
  const float* arr = ar + (size_t)m * CF + k * FD;
  float sl = 0.0f, sr = 0.0f;
#pragma unroll
  for (int f = 0; f < FD; ++f) { float zv = zr[f]; sl += zv * alr[f]; sr += zv * arr[f]; }
  el[i] = sl; er[i] = sr;
}

// ---------------- edge softmax (3 passes) ----------------

__global__ void k_edge_max(const int* __restrict__ src, const int* __restrict__ dst,
                           const float* __restrict__ el, const float* __restrict__ er,
                           float* __restrict__ emax, int N, int E, int total) {
  int i = blockIdx.x * blockDim.x + threadIdx.x;
  if (i >= total) return;
  int k = i & (KH - 1);
  int me = i >> 3;
  int e = me % E, m = me / E;
  int s = src[(size_t)m * E + e], d = dst[(size_t)m * E + e];
  float sc = el[((size_t)m * N + s) * KH + k] + er[((size_t)m * N + d) * KH + k];
  sc = sc > 0.0f ? sc : SLOPE * sc;
  atomicMaxF(&emax[((size_t)m * N + d) * KH + k], sc);
}

__global__ void k_edge_exp(const int* __restrict__ src, const int* __restrict__ dst,
                           const float* __restrict__ el, const float* __restrict__ er,
                           const float* __restrict__ emax, float* __restrict__ exB,
                           float* __restrict__ denom, int N, int E, int total) {
  int i = blockIdx.x * blockDim.x + threadIdx.x;
  if (i >= total) return;
  int k = i & (KH - 1);
  int me = i >> 3;
  int e = me % E, m = me / E;
  int s = src[(size_t)m * E + e], d = dst[(size_t)m * E + e];
  float sc = el[((size_t)m * N + s) * KH + k] + er[((size_t)m * N + d) * KH + k];
  sc = sc > 0.0f ? sc : SLOPE * sc;
  float ex = expf(sc - emax[((size_t)m * N + d) * KH + k]);
  exB[i] = ex;
  atomicAdd(&denom[((size_t)m * N + d) * KH + k], ex);
}

__global__ void k_edge_agg(const int* __restrict__ src, const int* __restrict__ dst,
                           const float* __restrict__ exB, const float* __restrict__ denom,
                           const float* __restrict__ z, float* __restrict__ outAgg,
                           int N, int E, int total) {
  int i = blockIdx.x * blockDim.x + threadIdx.x;
  if (i >= total) return;
  int f  = i & (FD - 1);
  int t  = i >> 5;                 // (m, e, k)
  int k  = t & (KH - 1);
  int me = t >> 3;
  int e = me % E, m = me / E;
  int s = src[(size_t)m * E + e], d = dst[(size_t)m * E + e];
  float alpha = exB[t] / (denom[((size_t)m * N + d) * KH + k] + 1e-9f);
  float val = alpha * z[((size_t)m * N + s) * CF + k * FD + f];
  atomicAdd(&outAgg[((size_t)m * N + d) * CF + k * FD + f], val);
}

// ---------------- bias + ELU, emit f32 + f16 copies ----------------

__global__ void k_finalize(float* __restrict__ outAgg, const float* __restrict__ bias,
                           _Float16* __restrict__ zp16, int N, int total) {
  int i = blockIdx.x * blockDim.x + threadIdx.x;
  if (i >= total) return;
  int m = i / (N * CF);
  int c = i % CF;
  float v = outAgg[i] + bias[(size_t)m * CF + c];
  v = v > 0.0f ? v : expf(v) - 1.0f;
  outAgg[i] = v;                     // zp (f32), consumed by combine
  zp16[i]   = (_Float16)v;           // f16 for the semantic WMMA GEMM
}

// ---------------- semantic GEMM: H = zp16 @ W1 (WMMA) ----------------
// rows = M*N (30000, multiple of 16); A tile staged in LDS; 8 waves x 16 cols = HID

__global__ __launch_bounds__(256) void k_sem_gemm(
    const _Float16* __restrict__ zp16, const _Float16* __restrict__ w1T16,
    float* __restrict__ H) {
  __shared__ _Float16 sA[16 * SAPAD];

  const int rowBase = blockIdx.x * 16;
  const int wave = threadIdx.x >> 5;
  const int lane = threadIdx.x & 31;
  const int r = lane & 15;
  const int hs = lane >> 4;

  stage_a_tile(zp16 + (size_t)rowBase * DIN, sA, threadIdx.x);
  __syncthreads();

  const _Float16* A = sA + r * SAPAD;
  const _Float16* B = w1T16 + (size_t)(wave * 16 + r) * DIN + hs * 16;
  v8f acc = {};
#pragma unroll
  for (int kk = 0; kk < DIN / 32; ++kk) {
    const int kb = kk * 32;
    v16h a = ld_frag(A + kb + hs * 16, A + kb + hs * 16 + 8);
    v16h b = ld_frag(B + kb, B + kb + 8);
    acc = wmma32f16(a, b, acc);
  }
#pragma unroll
  for (int j = 0; j < 8; ++j) {
    int row = rowBase + j + 8 * hs;
    H[(size_t)row * HID + wave * 16 + r] = acc[j];
  }
}

// per-block LDS reduction (ds_add_f32), then <=MPATH global atomics per block
__global__ void k_semw(const float* __restrict__ H, const float* __restrict__ b1,
                       const float* __restrict__ W2, float* __restrict__ wsum,
                       int N, int total) {
  __shared__ float part[MPATH];
  if (threadIdx.x < MPATH) part[threadIdx.x] = 0.0f;
  __syncthreads();

  int i = blockIdx.x * blockDim.x + threadIdx.x;
  if (i < total) {
    int m = i / N;
    const float* hr = H + (size_t)i * HID;
    float acc = 0.0f;
#pragma unroll 4
    for (int h = 0; h < HID; ++h) acc += tanhf(hr[h] + b1[h]) * W2[h];
    atomicAdd(&part[m], acc);          // LDS atomic
  }
  __syncthreads();
  if (threadIdx.x < MPATH) {
    float v = part[threadIdx.x];
    if (v != 0.0f) atomicAdd(&wsum[threadIdx.x], v);
  }
}

__global__ void k_beta(const float* __restrict__ wsum, float* __restrict__ beta, int N) {
  if (threadIdx.x == 0 && blockIdx.x == 0) {
    float w[MPATH], mx = -1e30f, s = 0.0f;
    for (int m = 0; m < MPATH; ++m) { w[m] = wsum[m] / (float)N; mx = fmaxf(mx, w[m]); }
    for (int m = 0; m < MPATH; ++m) { w[m] = expf(w[m] - mx); s += w[m]; }
    for (int m = 0; m < MPATH; ++m) beta[m] = w[m] / s;
  }
}

__global__ void k_combine(const float* __restrict__ zpf, const float* __restrict__ beta,
                          float* __restrict__ out, int N, int total) {
  int i = blockIdx.x * blockDim.x + threadIdx.x;
  if (i >= total) return;
  float b0 = beta[0], b1 = beta[1], b2 = beta[2];
  size_t stride = (size_t)N * CF;
  out[i] = b0 * zpf[i] + b1 * zpf[stride + i] + b2 * zpf[2 * stride + i];
}

// ---------------- host launcher ----------------

static inline size_t align_up(size_t x) { return (x + 255) & ~(size_t)255; }

extern "C" void kernel_launch(void* const* d_in, const int* in_sizes, int n_in,
                              void* d_out, int out_size, void* d_ws, size_t ws_size,
                              hipStream_t stream) {
  const float* h    = (const float*)d_in[0];
  const float* W    = (const float*)d_in[1];
  const float* al   = (const float*)d_in[2];
  const float* ar   = (const float*)d_in[3];
  const float* bias = (const float*)d_in[4];
  const float* W1   = (const float*)d_in[5];
  const float* b1   = (const float*)d_in[6];
  const float* W2   = (const float*)d_in[7];
  const int*   src  = (const int*)d_in[8];
  const int*   dst  = (const int*)d_in[9];
  float*       out  = (float*)d_out;

  const int N = in_sizes[0] / DIN;        // 10000 (multiple of 16)
  const int E = in_sizes[8] / MPATH;      // 200000

  // bump-allocated workspace (with aliasing of dead regions)
  char* p = (char*)d_ws;
  auto take = [&](size_t bytes) -> void* { void* q = (void*)p; p += align_up(bytes); return q; };
  _Float16* h16    = (_Float16*)take((size_t)N * DIN * 2);
  _Float16* wT16   = (_Float16*)take((size_t)MPATH * CF * DIN * 2);
  _Float16* w1T16  = (_Float16*)take((size_t)HID * DIN * 2);
  float*    z      = (float*)take((size_t)MPATH * N * CF * 4);
  float*    el     = (float*)take((size_t)MPATH * N * KH * 4);
  float*    er     = (float*)take((size_t)MPATH * N * KH * 4);
  float*    emax   = (float*)take((size_t)MPATH * N * KH * 4);
  float*    denom  = (float*)take((size_t)MPATH * N * KH * 4);
  float*    exB    = (float*)take((size_t)MPATH * E * KH * 4);
  float*    outAgg = (float*)take((size_t)MPATH * N * CF * 4);
  float*    wsum   = (float*)take(MPATH * 4);
  float*    beta   = (float*)take(MPATH * 4);
  // aliases: z dead after k_edge_agg -> reuse for zp16; exB dead -> reuse for H
  _Float16* zp16 = (_Float16*)z;                 // M*N*CF*2  <= M*N*CF*4
  float*    Hs   = exB;                          // M*N*HID*4 <= M*E*KH*4

  const int BLK = 256;
  auto nb = [](long long total, int blk) { return (unsigned)((total + blk - 1) / blk); };

  const long long mnc = (long long)MPATH * N * CF;
  const long long mnk = (long long)MPATH * N * KH;
  const long long mek = (long long)MPATH * E * KH;

  k_cvt_h<<<nb((long long)N * DIN, BLK), BLK, 0, stream>>>(h, h16, N * DIN);
  k_cvt_w<<<nb((long long)MPATH * CF * DIN, BLK), BLK, 0, stream>>>(W, wT16, MPATH * CF * DIN);
  k_cvt_w1<<<nb((long long)HID * DIN, BLK), BLK, 0, stream>>>(W1, w1T16, HID * DIN);
  k_init<<<nb(mnc, BLK), BLK, 0, stream>>>(emax, denom, outAgg, wsum, (int)mnk, (int)mnc);

  k_proj_gemm<<<dim3(N / 16, MPATH), BLK, 0, stream>>>(h16, wT16, z, N);
  k_eler<<<nb(mnk, BLK), BLK, 0, stream>>>(z, al, ar, el, er, N, (int)mnk);

  k_edge_max<<<nb(mek, BLK), BLK, 0, stream>>>(src, dst, el, er, emax, N, E, (int)mek);
  k_edge_exp<<<nb(mek, BLK), BLK, 0, stream>>>(src, dst, el, er, emax, exB, denom, N, E, (int)mek);
  k_edge_agg<<<nb(mek * FD, BLK), BLK, 0, stream>>>(src, dst, exB, denom, z, outAgg, N, E, (int)(mek * FD));

  k_finalize<<<nb(mnc, BLK), BLK, 0, stream>>>(outAgg, bias, zp16, N, (int)mnc);

  k_sem_gemm<<<nb((long long)MPATH * N, 16), BLK, 0, stream>>>(zp16, w1T16, Hs);
  k_semw<<<nb((long long)MPATH * N, BLK), BLK, 0, stream>>>(Hs, b1, W2, wsum, N, MPATH * N);
  k_beta<<<1, 32, 0, stream>>>(wsum, beta, N);
  k_combine<<<nb((long long)N * CF, BLK), BLK, 0, stream>>>(outAgg, beta, out, N, N * CF);
}
